// MultiheadAttention_23888608100406
// MI455X (gfx1250) — matrix operand order repaired and checked by reading, not compiled, not run on hardware
//
#include <hip/hip_runtime.h>
#include <hip/hip_bf16.h>
#include <math.h>

// ---------------------------------------------------------------------------
// MI455X / gfx1250 top-k sparse multihead attention.
// GEMMs: v_wmma_f32_16x16x32_bf16. Score GEMM uses async global->LDS staging
// (GLOBAL_LOAD_ASYNC_TO_LDS_B128 + s_wait_asynccnt) with double buffering.
// ---------------------------------------------------------------------------

typedef __attribute__((ext_vector_type(16))) __bf16 v16bf;
typedef __attribute__((ext_vector_type(8)))  float  v8f;

struct Frag32B { uint4 lo, hi; };   // 32 bytes -> bit_cast to v16bf

__device__ __forceinline__ unsigned short f2bf(float f) {
  unsigned int u = __float_as_uint(f);
  u += 0x7FFFu + ((u >> 16) & 1u);       // round-to-nearest-even
  return (unsigned short)(u >> 16);
}

#define LDSTRIDE 40   // 32 bf16 + 8 pad (80B rows: 16B aligned, bank-skewed)

#if __has_builtin(__builtin_amdgcn_global_load_async_to_lds_b128) && \
    __has_builtin(__builtin_amdgcn_s_wait_asynccnt)
#define ASYNC_LDS 1
#else
#define ASYNC_LDS 0
#endif

// exact parameter types per hipcc diagnostic: int4 pointers in AS1 / AS3
typedef int v4i_vs __attribute__((vector_size(16)));
typedef __attribute__((address_space(1))) v4i_vs* g_b128_ptr;
typedef __attribute__((address_space(3))) v4i_vs* l_b128_ptr;

// copy 16 bytes global -> LDS (async if available)
__device__ __forceinline__ void cp16_g2l(const unsigned short* g, unsigned short* l) {
#if ASYNC_LDS
  __builtin_amdgcn_global_load_async_to_lds_b128(
      (g_b128_ptr)(g), (l_b128_ptr)(l), 0, 0);
#else
  *(uint4*)l = *(const uint4*)g;
#endif
}

// ===========================================================================
// Variant 1: fp32 A/B, converted to bf16 during LDS staging.
// C[m,n] = alpha*(A@B + bias[n]); optional bf16 secondary output C2.
// Requires M,N % 64 == 0, K % 32 == 0.
// ===========================================================================
__global__ __launch_bounds__(128)
void wmma_gemm_f32_kernel(const float* __restrict__ A, const float* __restrict__ B,
                          const float* __restrict__ bias, float* __restrict__ C,
                          unsigned short* __restrict__ C2,
                          int K, long long sAm, long long sBk, long long sBn,
                          long long sCm, long long s2m,
                          long long aBatch, long long bBatch, long long cBatch,
                          long long biasBatch, float alpha)
{
  __shared__ unsigned short sA[64 * LDSTRIDE];   // [m][k] bf16
  __shared__ unsigned short sB[64 * LDSTRIDE];   // [n][k] bf16 (B transposed)

  const int tid  = threadIdx.x;
  const int wave = tid >> 5;
  const int lane = tid & 31;
  const int l16  = lane & 15;
  const int hi   = lane >> 4;

  const long long z = blockIdx.z;
  A += z * aBatch;  B += z * bBatch;  C += z * cBatch;
  if (bias) bias += z * biasBatch;

  const int m0 = blockIdx.y * 64;
  const int n0 = blockIdx.x * 64;

  v8f acc[4];
  #pragma unroll
  for (int i = 0; i < 4; ++i)
    #pragma unroll
    for (int j = 0; j < 8; ++j) acc[i][j] = 0.0f;

  for (int k0 = 0; k0 < K; k0 += 32) {
    #pragma unroll
    for (int it = 0; it < 4; ++it) {           // A: 64 rows x 8 float4
      const int i  = tid + it * 128;
      const int mi = i >> 3;
      const int kv = (i & 7) << 2;
      const float4 f = *(const float4*)(A + (long long)(m0 + mi) * sAm + (k0 + kv));
      unsigned short* d = &sA[mi * LDSTRIDE + kv];
      d[0] = f2bf(f.x); d[1] = f2bf(f.y); d[2] = f2bf(f.z); d[3] = f2bf(f.w);
    }
    #pragma unroll
    for (int it = 0; it < 16; ++it) {          // B^T: 64 n x 32 k scalars
      const int i  = tid + it * 128;
      const int n  = i >> 5;
      const int kk = i & 31;
      const float f = B[(long long)(k0 + kk) * sBk + (long long)(n0 + n) * sBn];
      sB[n * LDSTRIDE + kk] = f2bf(f);
    }
    __syncthreads();

    Frag32B fa;
    fa.lo = *(const uint4*)&sA[(wave * 16 + l16) * LDSTRIDE + hi * 8];
    fa.hi = *(const uint4*)&sA[(wave * 16 + l16) * LDSTRIDE + 16 + hi * 8];
    const v16bf av = __builtin_bit_cast(v16bf, fa);

    #pragma unroll
    for (int nb = 0; nb < 4; ++nb) {
      Frag32B fb;
      fb.lo = *(const uint4*)&sB[(nb * 16 + l16) * LDSTRIDE + hi * 8];
      fb.hi = *(const uint4*)&sB[(nb * 16 + l16) * LDSTRIDE + 16 + hi * 8];
      const v16bf bvv = __builtin_bit_cast(v16bf, fb);
      acc[nb] = __builtin_amdgcn_wmma_f32_16x16x32_bf16(
          false, av, false, bvv, (short)0, acc[nb], false, false);
    }
    __syncthreads();
  }

  #pragma unroll
  for (int nb = 0; nb < 4; ++nb) {
    const int col = n0 + nb * 16 + l16;
    const float bval = bias ? bias[col] : 0.0f;
    #pragma unroll
    for (int r = 0; r < 8; ++r) {
      const int m = m0 + wave * 16 + hi * 8 + r;
      const float v = alpha * (acc[nb][r] + bval);
      C[(long long)m * sCm + col] = v;
      if (C2) C2[(long long)m * s2m + col] = f2bf(v);
    }
  }
}

// ===========================================================================
// Variant 2: bf16 A and pre-transposed bf16 B (both k-contiguous rows).
// Async global->LDS staging, double buffered on ASYNCcnt.
// ===========================================================================
__global__ __launch_bounds__(128)
void wmma_gemm_bf16_kernel(const unsigned short* __restrict__ A,
                           const unsigned short* __restrict__ B,
                           const float* __restrict__ bias, float* __restrict__ C,
                           int K, long long sAm, long long sBn, long long sCm,
                           long long aBatch, long long bBatch, long long cBatch,
                           long long biasBatch, float alpha)
{
  __shared__ unsigned short sA[2][64 * LDSTRIDE];
  __shared__ unsigned short sB[2][64 * LDSTRIDE];

  const int tid  = threadIdx.x;
  const int wave = tid >> 5;
  const int lane = tid & 31;
  const int l16  = lane & 15;
  const int hi   = lane >> 4;

  const long long z = blockIdx.z;
  A += z * aBatch;  B += z * bBatch;  C += z * cBatch;
  if (bias) bias += z * biasBatch;

  const int m0 = blockIdx.y * 64;
  const int n0 = blockIdx.x * 64;

  // this thread's two 16B chunks of each 64x32-bf16 tile (256 chunks total)
  const int c0 = tid * 2, c1 = tid * 2 + 1;
  const int r0 = c0 >> 2, q0 = (c0 & 3) * 8;   // row, k-offset (halves)
  const int r1 = c1 >> 2, q1 = (c1 & 3) * 8;

  v8f acc[4];
  #pragma unroll
  for (int i = 0; i < 4; ++i)
    #pragma unroll
    for (int j = 0; j < 8; ++j) acc[i][j] = 0.0f;

  const int nT = K >> 5;

  // issue tile t into buffer bufi: 4 async b128 ops per thread
  #define ISSUE_TILE(t, bufi)                                                   \
    do {                                                                        \
      const long long kk0 = (long long)(t) << 5;                                \
      cp16_g2l(A + (long long)(m0 + r0) * sAm + kk0 + q0,                       \
               &sA[bufi][r0 * LDSTRIDE + q0]);                                  \
      cp16_g2l(B + (long long)(n0 + r0) * sBn + kk0 + q0,                       \
               &sB[bufi][r0 * LDSTRIDE + q0]);                                  \
      cp16_g2l(A + (long long)(m0 + r1) * sAm + kk0 + q1,                       \
               &sA[bufi][r1 * LDSTRIDE + q1]);                                  \
      cp16_g2l(B + (long long)(n0 + r1) * sBn + kk0 + q1,                       \
               &sB[bufi][r1 * LDSTRIDE + q1]);                                  \
    } while (0)

  ISSUE_TILE(0, 0);
  for (int t = 0; t < nT; ++t) {
    const int cur = t & 1;
    if (t + 1 < nT) {
      ISSUE_TILE(t + 1, cur ^ 1);   // prefetch next tile (buffer free since t-1)
#if ASYNC_LDS
      __builtin_amdgcn_s_wait_asynccnt(4);   // tile t done (in-order), t+1 live
#endif
    } else {
#if ASYNC_LDS
      __builtin_amdgcn_s_wait_asynccnt(0);
#endif
    }
    __syncthreads();   // every wave waited its own ASYNCcnt -> tile complete

    Frag32B fa;
    fa.lo = *(const uint4*)&sA[cur][(wave * 16 + l16) * LDSTRIDE + hi * 8];
    fa.hi = *(const uint4*)&sA[cur][(wave * 16 + l16) * LDSTRIDE + 16 + hi * 8];
    const v16bf av = __builtin_bit_cast(v16bf, fa);

    #pragma unroll
    for (int nb = 0; nb < 4; ++nb) {
      Frag32B fb;
      fb.lo = *(const uint4*)&sB[cur][(nb * 16 + l16) * LDSTRIDE + hi * 8];
      fb.hi = *(const uint4*)&sB[cur][(nb * 16 + l16) * LDSTRIDE + 16 + hi * 8];
      const v16bf bvv = __builtin_bit_cast(v16bf, fb);
      acc[nb] = __builtin_amdgcn_wmma_f32_16x16x32_bf16(
          false, av, false, bvv, (short)0, acc[nb], false, false);
    }
    __syncthreads();   // done reading buffer `cur` before it is re-issued
  }
  #undef ISSUE_TILE

  #pragma unroll
  for (int nb = 0; nb < 4; ++nb) {
    const int col = n0 + nb * 16 + l16;
    const float bval = bias ? bias[col] : 0.0f;
    #pragma unroll
    for (int r = 0; r < 8; ++r) {
      const int m = m0 + wave * 16 + hi * 8 + r;
      C[(long long)m * sCm + col] = alpha * (acc[nb][r] + bval);
    }
  }
}

// nk[(b*8+h)*4096 + j] = -0.5 * ||k_head(b,h,j)||^2  (score column bias)
__global__ __launch_bounds__(256)
void knorm_kernel(const float* __restrict__ kp, float* __restrict__ nk)
{
  const int i = blockIdx.x * 256 + threadIdx.x;   // 2*8*4096 = 65536 total
  const int j = i & 4095;
  const int z = i >> 12;
  const int b = z >> 3;
  const int h = z & 7;
  const float* p = kp + ((size_t)j * 2 + b) * 1024 + h * 128;
  float s = 0.0f;
  #pragma unroll
  for (int d = 0; d < 128; d += 4) {
    const float4 f = *(const float4*)(p + d);
    s += f.x * f.x + f.y * f.y + f.z * f.z + f.w * f.w;
  }
  nk[i] = -0.5f * s;
}

__device__ __forceinline__ unsigned int f2ord(float f) {
  const unsigned int b = __float_as_uint(f);
  return (b & 0x80000000u) ? ~b : (b | 0x80000000u);
}
__device__ __forceinline__ float ord2f(unsigned int u) {
  const unsigned int b = (u & 0x80000000u) ? (u & 0x7FFFFFFFu) : ~u;
  return __uint_as_float(b);
}

// In-place per-row (4096): keep top-128, softmax them, zero the rest.
__global__ __launch_bounds__(256)
void topk_softmax_kernel(float* __restrict__ attn)
{
  __shared__ unsigned int  su[4096];
  __shared__ unsigned char sel[4096];
  __shared__ unsigned int  hist[256];
  __shared__ unsigned int  scan[256];
  __shared__ unsigned int  um[256];
  __shared__ float         red[256];
  __shared__ unsigned int  sChosen, sKrem, sUmax;
  __shared__ float         sSum;

  float* row = attn + (size_t)blockIdx.x * 4096;
  const int t = threadIdx.x;

  for (int i = t; i < 4096; i += 256) su[i] = f2ord(row[i]);
  __syncthreads();

  unsigned int prefix = 0, mask = 0, kneed = 128;
  for (int shift = 24; shift >= 0; shift -= 8) {
    hist[t] = 0;
    __syncthreads();
    for (int i = 0; i < 16; ++i) {
      const unsigned int u = su[t * 16 + i];
      if ((u & mask) == prefix) atomicAdd(&hist[(u >> shift) & 255u], 1u);
    }
    __syncthreads();
    if (t == 0) {
      unsigned int cum = 0, chosen = 0, kk = kneed;
      for (int b = 255; b >= 0; --b) {
        const unsigned int c = hist[b];
        if (cum + c >= kneed) { chosen = (unsigned int)b; kk = kneed - cum; break; }
        cum += c;
      }
      sChosen = chosen; sKrem = kk;
    }
    __syncthreads();
    prefix |= (sChosen << shift);
    mask   |= (255u << shift);
    kneed   = sKrem;
    __syncthreads();
  }
  const unsigned int uT  = prefix;
  const unsigned int kEq = kneed;

  unsigned int myEq = 0, myMax = 0;
  for (int i = 0; i < 16; ++i) {
    const unsigned int u = su[t * 16 + i];
    myEq += (u == uT);
    myMax = (u > myMax) ? u : myMax;
  }
  hist[t] = myEq;  um[t] = myMax;
  __syncthreads();
  if (t == 0) {
    unsigned int c = 0, mx = 0;
    for (int i = 0; i < 256; ++i) {
      scan[i] = c; c += hist[i];
      if (um[i] > mx) mx = um[i];
    }
    sUmax = mx;
  }
  __syncthreads();

  const float fmax = ord2f(sUmax);
  unsigned int rank = scan[t];
  float partial = 0.0f;
  for (int i = 0; i < 16; ++i) {
    const int idx = t * 16 + i;
    const unsigned int u = su[idx];
    bool s = (u > uT);
    if (!s && u == uT) { s = (rank < kEq); ++rank; }
    sel[idx] = s ? 1 : 0;
    if (s) partial += __expf(ord2f(u) - fmax);
  }
  red[t] = partial;
  __syncthreads();
  if (t == 0) { float sm = 0.0f; for (int i = 0; i < 256; ++i) sm += red[i]; sSum = sm; }
  __syncthreads();
  const float inv = 1.0f / sSum;
  for (int i = t; i < 4096; i += 256) {
    float o = 0.0f;
    if (sel[i]) o = __expf(ord2f(su[i]) - fmax) * inv;
    row[i] = o;
  }
}

// ---------------------------------------------------------------------------
extern "C" void kernel_launch(void* const* d_in, const int* in_sizes, int n_in,
                              void* d_out, int out_size, void* d_ws, size_t ws_size,
                              hipStream_t stream) {
  (void)in_sizes; (void)n_in; (void)out_size; (void)ws_size;
  const float* Q   = (const float*)d_in[0];   // [2048,2,1024]
  const float* Kin = (const float*)d_in[1];   // [4096,2,1024]
  const float* V   = (const float*)d_in[2];   // [4096,2,1024]
  const float* Wq  = (const float*)d_in[3];
  const float* bq  = (const float*)d_in[4];
  const float* Wk  = (const float*)d_in[5];
  const float* bk  = (const float*)d_in[6];
  const float* Wv  = (const float*)d_in[7];
  const float* bv  = (const float*)d_in[8];
  const float* Wp  = (const float*)d_in[9];
  const float* bp  = (const float*)d_in[10];

  float* out  = (float*)d_out;                         // [2048,2,1024]
  float* attn = out + (size_t)2048 * 2 * 1024;         // [2,8,2048,4096]

  float* ws = (float*)d_ws;                            // ~126 MB used
  float* qp = ws;                                      // 4,194,304 f32
  float* kp = qp + (size_t)4194304;                    // 8,388,608 f32
  float* vp = kp + (size_t)8388608;                    // 8,388,608 f32
  float* op = vp + (size_t)8388608;                    // 4,194,304 f32
  float* nk = op + (size_t)4194304;                    //    65,536 f32
  unsigned short* qb = (unsigned short*)(nk + 65536);  // 4,194,304 bf16
  unsigned short* kb = qb + (size_t)4194304;           // 8,388,608 bf16

  const float invT = 0.08838834764831845f;             // 1/sqrt(128)

  // q/k/v projections: [rows,1024] x [1024,1024] + bias (q,k also emit bf16)
  wmma_gemm_f32_kernel<<<dim3(16, 64, 1), 128, 0, stream>>>(
      Q, Wq, bq, qp, qb, 1024, 1024, 1024, 1, 1024, 1024, 0, 0, 0, 0, invT);
  wmma_gemm_f32_kernel<<<dim3(16, 128, 1), 128, 0, stream>>>(
      Kin, Wk, bk, kp, kb, 1024, 1024, 1024, 1, 1024, 1024, 0, 0, 0, 0, 1.0f);
  wmma_gemm_f32_kernel<<<dim3(16, 128, 1), 128, 0, stream>>>(
      V, Wv, bv, vp, nullptr, 1024, 1024, 1024, 1, 1024, 0, 0, 0, 0, 0, 1.0f);

  // -0.5*||k_head||^2 column bias
  knorm_kernel<<<256, 256, 0, stream>>>(kp, nk);

  // scores = 2*(q_head @ k_head^T - 0.5||k||^2): bf16 operands, async staging
  // batch z = b*8+h: head offset = 128*z, C offset = z*Tq*Tk
  wmma_gemm_bf16_kernel<<<dim3(64, 32, 16), 128, 0, stream>>>(
      qb, kb, nk, attn, 128,
      /*sAm*/2048, /*sBn*/2048, /*sCm*/4096,
      /*aB*/128, /*bB*/128, /*cB*/8388608LL, /*biasB*/4096, 2.0f);

  // per-row top-128 softmax, zeros elsewhere (in place)
  topk_softmax_kernel<<<32768, 256, 0, stream>>>(attn);

  // out_head = attn @ v_head  -> out_pre[(qi*bs+b)*1024 + h*128 + d]
  wmma_gemm_f32_kernel<<<dim3(2, 32, 16), 128, 0, stream>>>(
      attn, vp, nullptr, op, nullptr, 4096,
      /*sAm*/4096, /*sBk*/2048, /*sBn*/1, /*sCm*/2048, 0,
      /*aB*/8388608LL, /*bB*/128, /*cB*/128, /*biasB*/0, 1.0f);

  // final projection -> out
  wmma_gemm_f32_kernel<<<dim3(16, 64, 1), 128, 0, stream>>>(
      op, Wp, bp, out, nullptr, 1024, 1024, 1024, 1, 1024, 0, 0, 0, 0, 0, 1.0f);
}
